// MultiHeadAttention_47923245089132
// MI455X (gfx1250) — matrix operand order repaired and checked by reading, not compiled, not run on hardware
//
#include <hip/hip_runtime.h>

// ---------------------------------------------------------------------------
// MI455X (gfx1250) multi-head attention, bf16 WMMA path with fp32 accumulate.
// D_MODEL=1024, H=16, d_k=64, B=2, S=2048.
// ---------------------------------------------------------------------------

typedef __bf16 bhalf;
typedef __attribute__((ext_vector_type(16))) __bf16 v16bf;
typedef __attribute__((ext_vector_type(8)))  __bf16 v8bf;
typedef __attribute__((ext_vector_type(8)))  float  v8f;

union FragBF { v16bf v; v8bf h[2]; };

#define D_MODEL 1024
#define N_HEADS 16
#define D_K     64
#define BATCH   2
#define SEQ     2048
#define ROWS    (BATCH * SEQ)   // 4096

__device__ __forceinline__ bhalf f2bf(float f) {
  unsigned u = __builtin_bit_cast(unsigned, f);
  u += 0x7FFFu + ((u >> 16) & 1u);              // round-to-nearest-even
  unsigned short hv = (unsigned short)(u >> 16);
  return __builtin_bit_cast(bhalf, hv);
}

__device__ __forceinline__ v8f wmma_bf16(const FragBF& a, const FragBF& b, v8f c) {
  // D(16x16,f32) = A(16x32,bf16) * B(32x16,bf16) + C
  return __builtin_amdgcn_wmma_f32_16x16x32_bf16(
      /*neg_a=*/false, a.v, /*neg_b=*/false, b.v,
      /*c_mod=*/(short)0, c, /*reuse_a=*/false, /*reuse_b=*/false);
}

// ---------------------------------------------------------------------------
// Weight transpose + fp32 -> bf16:  W[K][N] -> Wt[N][K]
// ---------------------------------------------------------------------------
__global__ __launch_bounds__(256) void transpose_w_kernel(const float* __restrict__ W,
                                                          bhalf* __restrict__ Wt) {
  int idx = blockIdx.x * 256 + threadIdx.x;     // 1M elements
  int k = idx >> 10;
  int n = idx & 1023;
  Wt[(size_t)n * D_MODEL + k] = f2bf(W[idx]);
}

// ---------------------------------------------------------------------------
// GEMM: C[M=4096][N=1024] = A[M][K=1024] * Wt^T + bias
//   MODE 0: A fp32, out bf16 scattered to Qh/Kh head layout [b][h][s][d]
//   MODE 1: A fp32, out bf16 scattered to Vt layout          [b][h][d][s]
//   MODE 2: A bf16, out fp32 row-major [m][n]  (final projection)
// Block 256 thr = 8 waves as 4(M)x2(N); wave tile 32x32; block tile 128x64.
// ---------------------------------------------------------------------------
template <int MODE>
__global__ __launch_bounds__(256) void gemm_wmma_kernel(const void* __restrict__ A_,
                                                        const bhalf* __restrict__ Wt,
                                                        const float* __restrict__ bias,
                                                        void* __restrict__ C_) {
  constexpr int K = D_MODEL, N = D_MODEL;
  const int lane = threadIdx.x & 31;
  const int wave = threadIdx.x >> 5;
  const int wm = wave >> 1, wn = wave & 1;
  const int tileM = blockIdx.y * 128 + wm * 32;
  const int tileN = blockIdx.x * 64 + wn * 32;
  const int half = lane >> 4;     // which 16-lane group
  const int l16  = lane & 15;

  v8f acc[2][2] = {};

  for (int k0 = 0; k0 < K; k0 += 32) {
    FragBF a[2], b[2];
#pragma unroll
    for (int mi = 0; mi < 2; ++mi) {
      const int row = tileM + mi * 16 + l16;
      if (MODE < 2) {
        const float* ap = (const float*)A_ + (size_t)row * K + k0 + half * 8;
        v8f c0 = *(const v8f*)ap;
        v8f c1 = *(const v8f*)(ap + 16);
        if (k0 + 64 < K) __builtin_prefetch(ap + 32, 0, 1);
#pragma unroll
        for (int e = 0; e < 8; ++e) { a[mi].h[0][e] = f2bf(c0[e]); a[mi].h[1][e] = f2bf(c1[e]); }
      } else {
        const bhalf* ap = (const bhalf*)A_ + (size_t)row * K + k0 + half * 8;
        a[mi].h[0] = *(const v8bf*)ap;
        a[mi].h[1] = *(const v8bf*)(ap + 16);
        if (k0 + 64 < K) __builtin_prefetch(ap + 32, 0, 1);
      }
    }
#pragma unroll
    for (int ni = 0; ni < 2; ++ni) {
      const int col = tileN + ni * 16 + l16;
      const bhalf* bp = Wt + (size_t)col * K + k0 + half * 8;
      b[ni].h[0] = *(const v8bf*)bp;
      b[ni].h[1] = *(const v8bf*)(bp + 16);
    }
#pragma unroll
    for (int mi = 0; mi < 2; ++mi)
#pragma unroll
      for (int ni = 0; ni < 2; ++ni)
        acc[mi][ni] = wmma_bf16(a[mi], b[ni], acc[mi][ni]);
  }

  // Epilogue: C layout — vgpr j -> M = j + 8*half, N = l16.
#pragma unroll
  for (int mi = 0; mi < 2; ++mi) {
#pragma unroll
    for (int ni = 0; ni < 2; ++ni) {
      const int n  = tileN + ni * 16 + l16;
      const float bv = bias[n];
#pragma unroll
      for (int j = 0; j < 8; ++j) {
        const int m = tileM + mi * 16 + j + 8 * half;
        const float val = acc[mi][ni][j] + bv;
        if (MODE == 2) {
          ((float*)C_)[(size_t)m * N + n] = val;
        } else {
          const int bb = m >> 11, s = m & 2047;      // m = b*S + s
          const int hh = n >> 6,  d = n & 63;        // n = h*64 + d
          bhalf* out = (bhalf*)C_;
          if (MODE == 0)
            out[(((size_t)bb * N_HEADS + hh) * SEQ + s) * D_K + d] = f2bf(val);
          else
            out[(((size_t)bb * N_HEADS + hh) * D_K + d) * SEQ + s] = f2bf(val);
        }
      }
    }
  }
}

// ---------------------------------------------------------------------------
// Flash attention: each wave owns a 16-query tile of one (b,h); streams
// keys/values 32 at a time. 8 waves / block -> 512 blocks for 4096 tiles.
// ---------------------------------------------------------------------------
__global__ __launch_bounds__(256) void flash_attn_kernel(const bhalf* __restrict__ Qh,
                                                         const bhalf* __restrict__ Kh,
                                                         const bhalf* __restrict__ Vt,
                                                         bhalf* __restrict__ Aout) {
  __shared__ __align__(16) bhalf Pl[8][16][40];   // per-wave P bounce (padded)

  const int lane = threadIdx.x & 31;
  const int wave = threadIdx.x >> 5;
  const int half = lane >> 4;
  const int l16  = lane & 15;

  const int tile = blockIdx.x * 8 + wave;         // 0..4095
  const int bh = tile >> 7;                       // SEQ/16 = 128 tiles per head
  const int q0 = (tile & 127) << 4;
  const bhalf* Qb = Qh + (size_t)bh * SEQ * D_K;
  const bhalf* Kb = Kh + (size_t)bh * SEQ * D_K;
  const bhalf* Vb = Vt + (size_t)bh * D_K * SEQ;

  // Q fragments for d=[0,32) and d=[32,64), kept resident.
  FragBF qf[2];
#pragma unroll
  for (int kd = 0; kd < 2; ++kd) {
    const bhalf* qp = Qb + (size_t)(q0 + l16) * D_K + kd * 32 + half * 16;
    qf[kd].h[0] = *(const v8bf*)qp;
    qf[kd].h[1] = *(const v8bf*)(qp + 8);
  }

  v8f o[4] = {};                                  // 16x64 output accumulator
  float mrow[8], lrow[8];
#pragma unroll
  for (int j = 0; j < 8; ++j) { mrow[j] = -1e30f; lrow[j] = 0.0f; }

  const float scale = 0.125f;                     // 1/sqrt(64)

  for (int kb = 0; kb < SEQ; kb += 32) {
    // --- S = Q * K^T (two 16x16 tiles covering 32 key positions) ---
    v8f s0 = {}, s1 = {};
    {
      FragBF kf;
      const bhalf* kp = Kb + (size_t)(kb + l16) * D_K + half * 16;
      kf.h[0] = *(const v8bf*)kp;       kf.h[1] = *(const v8bf*)(kp + 8);
      s0 = wmma_bf16(qf[0], kf, s0);
      kf.h[0] = *(const v8bf*)(kp + 32); kf.h[1] = *(const v8bf*)(kp + 40);
      s0 = wmma_bf16(qf[1], kf, s0);
      const bhalf* kp1 = kp + (size_t)16 * D_K;
      kf.h[0] = *(const v8bf*)kp1;       kf.h[1] = *(const v8bf*)(kp1 + 8);
      s1 = wmma_bf16(qf[0], kf, s1);
      kf.h[0] = *(const v8bf*)(kp1 + 32); kf.h[1] = *(const v8bf*)(kp1 + 40);
      s1 = wmma_bf16(qf[1], kf, s1);
    }

    // --- Online softmax: row stats live in vgpr j (row = j + 8*half),
    //     replicated across the 16 lanes of each half via shfl reductions. ---
#pragma unroll
    for (int j = 0; j < 8; ++j) {
      float a0 = s0[j] * scale, a1 = s1[j] * scale;
      float rm = fmaxf(a0, a1);
#pragma unroll
      for (int off = 8; off > 0; off >>= 1) rm = fmaxf(rm, __shfl_xor(rm, off, 16));
      const float mn = fmaxf(mrow[j], rm);
      const float alpha = __expf(mrow[j] - mn);
      mrow[j] = mn;
      const float p0 = __expf(a0 - mn);
      const float p1 = __expf(a1 - mn);
      float rs = p0 + p1;
#pragma unroll
      for (int off = 8; off > 0; off >>= 1) rs += __shfl_xor(rs, off, 16);
      lrow[j] = lrow[j] * alpha + rs;
#pragma unroll
      for (int t = 0; t < 4; ++t) o[t][j] *= alpha;
      // Stash P (bf16) into LDS in C-layout coordinates.
      Pl[wave][j + 8 * half][l16]      = f2bf(p0);
      Pl[wave][j + 8 * half][16 + l16] = f2bf(p1);
    }

    // Wave-local LDS fence: stores above -> A-layout reads below.
    asm volatile("s_wait_dscnt 0" ::: "memory");

    // Re-read P as a 16x32 A-matrix fragment.
    FragBF pf;
    pf.h[0] = *(const v8bf*)&Pl[wave][l16][half * 8];
    pf.h[1] = *(const v8bf*)&Pl[wave][l16][16 + half * 8];
    asm volatile("" ::: "memory");

    // --- O += P * V (V stored column-major: Vt[d][s]) ---
#pragma unroll
    for (int t = 0; t < 4; ++t) {
      const bhalf* vp = Vb + (size_t)(t * 16 + l16) * SEQ + kb + half * 8;
      FragBF vf;
      vf.h[0] = *(const v8bf*)vp;
      vf.h[1] = *(const v8bf*)(vp + 16);
      o[t] = wmma_bf16(pf, vf, o[t]);
    }
  }

  // Normalize and store to Aout[b][s][h*64+d] (bf16, feeds final GEMM).
  const int bb = bh >> 4, hh = bh & 15;
#pragma unroll
  for (int j = 0; j < 8; ++j) {
    const float inv = 1.0f / lrow[j];
    const int s = q0 + j + 8 * half;
    const size_t rowbase = ((size_t)bb * SEQ + s) * D_MODEL + hh * D_K;
#pragma unroll
    for (int t = 0; t < 4; ++t)
      Aout[rowbase + t * 16 + l16] = f2bf(o[t][j] * inv);
  }
}

// ---------------------------------------------------------------------------
// Host-side launcher.
// Inputs: q,k,v, Wq,bq, Wk,bk, Wv,bv, Wo,bo  (all fp32)
// Workspace layout (40 MB total):
//   [0,8MB)   WtQ/WtK/WtV/WtO  (4 x 1Mi bf16)
//   [8,16MB)  Qh   [16,24MB) Kh   [24,32MB) Vt   [32,40MB) Aout
// ---------------------------------------------------------------------------
extern "C" void kernel_launch(void* const* d_in, const int* in_sizes, int n_in,
                              void* d_out, int out_size, void* d_ws, size_t ws_size,
                              hipStream_t stream) {
  (void)in_sizes; (void)n_in; (void)out_size; (void)ws_size;

  const float* q  = (const float*)d_in[0];
  const float* k  = (const float*)d_in[1];
  const float* v  = (const float*)d_in[2];
  const float* Wq = (const float*)d_in[3];
  const float* bq = (const float*)d_in[4];
  const float* Wk = (const float*)d_in[5];
  const float* bk = (const float*)d_in[6];
  const float* Wv = (const float*)d_in[7];
  const float* bv = (const float*)d_in[8];
  const float* Wo = (const float*)d_in[9];
  const float* bo = (const float*)d_in[10];

  char* ws = (char*)d_ws;
  const size_t MB = (size_t)1 << 20;
  bhalf* WtQ = (bhalf*)(ws + 0 * MB);
  bhalf* WtK = (bhalf*)(ws + 2 * MB);
  bhalf* WtV = (bhalf*)(ws + 4 * MB);
  bhalf* WtO = (bhalf*)(ws + 6 * MB);
  bhalf* Qh  = (bhalf*)(ws + 8 * MB);
  bhalf* Kh  = (bhalf*)(ws + 16 * MB);
  bhalf* Vt  = (bhalf*)(ws + 24 * MB);
  bhalf* Ao  = (bhalf*)(ws + 32 * MB);

  transpose_w_kernel<<<4096, 256, 0, stream>>>(Wq, WtQ);
  transpose_w_kernel<<<4096, 256, 0, stream>>>(Wk, WtK);
  transpose_w_kernel<<<4096, 256, 0, stream>>>(Wv, WtV);
  transpose_w_kernel<<<4096, 256, 0, stream>>>(Wo, WtO);

  dim3 gg(D_MODEL / 64, ROWS / 128);   // (16, 32)
  gemm_wmma_kernel<0><<<gg, 256, 0, stream>>>(q, WtQ, bq, Qh);
  gemm_wmma_kernel<0><<<gg, 256, 0, stream>>>(k, WtK, bk, Kh);
  gemm_wmma_kernel<1><<<gg, 256, 0, stream>>>(v, WtV, bv, Vt);

  flash_attn_kernel<<<ROWS * N_HEADS / 16 / 8, 256, 0, stream>>>(Qh, Kh, Vt, Ao);

  gemm_wmma_kernel<2><<<gg, 256, 0, stream>>>(Ao, WtO, bo, d_out);
}
